// VanillaRNNLM_27547920237226
// MI455X (gfx1250) — matrix operand order repaired and checked by reading, not compiled, not run on hardware
//
#include <hip/hip_runtime.h>
#include <math.h>

// ---------------------------------------------------------------------------
// VanillaRNNLM on MI455X (gfx1250, wave32, WMMA).
//   x[16,256] -> emb gather -> ih GEMM -> 256 sequential RNN steps -> proj GEMM
// All GEMMs use v_wmma_f32_16x16x32_bf16 (f32 accumulate).
// Weights are pre-transposed to K-major bf16; A fragments load as
// 16-byte vector loads, B fragments as single bf16 loads strided by N. Inner loops are software-pipelined one stage deep so
// WMMAs don't wait on their own fragment loads.
// ---------------------------------------------------------------------------

typedef __attribute__((ext_vector_type(16))) __bf16 v16bf;
typedef __attribute__((ext_vector_type(8)))  __bf16 v8bf;
typedef __attribute__((ext_vector_type(8)))  float  v8f;

static constexpr int BATCH = 16;
static constexpr int SEQ   = 256;
static constexpr int EMB   = 512;
static constexpr int HID   = 1024;
static constexpr int VOCAB = 32000;
static constexpr int MROWS = BATCH * SEQ;   // 4096

#define DEVI static __device__ __forceinline__

DEVI v16bf combine16(v8bf lo, v8bf hi) {
  v16bf r;
#pragma unroll
  for (int i = 0; i < 8; ++i) { r[i] = lo[i]; r[i + 8] = hi[i]; }
  return r;
}

// A fragment (16x32 bf16, MxK). Per ISA 7.12.2: lane&15 = M row,
// lanes>=16 hold K=8..15 / 24..31; halves i=0..7 -> K=khalf+0..7,
// i=8..15 -> K=khalf+16..23.  p = &Arow[k0 + khalf], khalf = (lane>>4)*8.
DEVI v16bf load_afrag(const __bf16* p) {
  return combine16(*(const v8bf*)p, *(const v8bf*)(p + 16));
}

// B fragment (32x16 bf16, KxN), B stored K-major: lane&15 = N col,
// i=0..7 -> K=khalf+0..7, i=8..15 -> K=khalf+16..23 (strided by ld).
// p = &BT[(k0+khalf)*ld + n0 + (lane&15)], khalf = (lane>>4)*8.
DEVI v16bf load_bfrag(const __bf16* p, long ld) {
  v16bf r;
#pragma unroll
  for (int i = 0; i < 8; ++i) { r[i] = p[i * ld]; r[i + 8] = p[(16 + i) * ld]; }
  return r;
}

DEVI v8f wmma_bf16(v16bf a, v16bf b, v8f c) {
  return __builtin_amdgcn_wmma_f32_16x16x32_bf16(false, a, false, b,
                                                 (short)0, c, false, false);
}

// ---------------------------------------------------------------------------
// Elementwise prep kernels
// ---------------------------------------------------------------------------

// e_bf16[m, k] = bf16(emb[x[m], k])      m = b*SEQ + t
__global__ void k_gather_embed(const int* __restrict__ x,
                               const float* __restrict__ emb,
                               __bf16* __restrict__ e, int total) {
  int idx = blockIdx.x * blockDim.x + threadIdx.x;
  if (idx >= total) return;
  int m = idx >> 9;          // / EMB
  int k = idx & (EMB - 1);
  e[idx] = (__bf16)emb[(long)x[m] * EMB + k];
}

// W[R, K] f32 row-major  ->  WT[K, R] bf16 (K-major, GEMM "BT" operand)
__global__ void k_transpose_bf16(const float* __restrict__ W,
                                 __bf16* __restrict__ WT,
                                 int R, int K) {
  long idx = (long)blockIdx.x * blockDim.x + threadIdx.x;
  long total = (long)R * K;
  if (idx >= total) return;
  int n = (int)(idx / K);
  int k = (int)(idx % K);
  WT[(long)k * R + n] = (__bf16)W[idx];
}

// ---------------------------------------------------------------------------
// Generic bf16 WMMA GEMM:  C[M,N] = A[M,K] * BT[K,N] + bias[N]   (f32 out)
// Block = 128 threads (4 waves). Each wave computes a 16x64 tile (4 WMMA tiles
// sharing one A fragment). Grid = (M/16, N/256). K must be a multiple of 32.
// ---------------------------------------------------------------------------
__global__ void k_gemm_bf16_wmma(const __bf16* __restrict__ A,
                                 const __bf16* __restrict__ BT,
                                 const float* __restrict__ bias,
                                 float* __restrict__ C,
                                 int M, int N, int K) {
  const int lane = threadIdx.x & 31;
  const int wave = threadIdx.x >> 5;
  const int m0 = blockIdx.x * 16;
  const int n0 = blockIdx.y * 256 + wave * 64;

  const int khalf = (lane >> 4) * 8;
  const __bf16* ap = A + (long)(m0 + (lane & 15)) * K + khalf;
  const __bf16* bp = BT + (long)khalf * N + n0 + (lane & 15);
  const long bstep = (long)32 * N;

  v8f acc[4] = {};

  // Prologue: fragments for k-chunk 0
  v16bf a  = load_afrag(ap);
  v16bf b0 = load_bfrag(bp, N);
  v16bf b1 = load_bfrag(bp + 16, N);
  v16bf b2 = load_bfrag(bp + 32, N);
  v16bf b3 = load_bfrag(bp + 48, N);

  for (int k0 = 32; k0 < K; k0 += 32) {
    ap += 32;
    bp += bstep;
    // CDNA5 global_prefetch_b8: start pulling chunk k0+32's B slab
    __builtin_prefetch(bp + bstep, 0, 3);
    // Issue next chunk's loads before consuming current fragments
    v16bf an  = load_afrag(ap);
    v16bf bn0 = load_bfrag(bp, N);
    v16bf bn1 = load_bfrag(bp + 16, N);
    v16bf bn2 = load_bfrag(bp + 32, N);
    v16bf bn3 = load_bfrag(bp + 48, N);

    acc[0] = wmma_bf16(a, b0, acc[0]);
    acc[1] = wmma_bf16(a, b1, acc[1]);
    acc[2] = wmma_bf16(a, b2, acc[2]);
    acc[3] = wmma_bf16(a, b3, acc[3]);

    a = an; b0 = bn0; b1 = bn1; b2 = bn2; b3 = bn3;
  }
  // Epilogue chunk
  acc[0] = wmma_bf16(a, b0, acc[0]);
  acc[1] = wmma_bf16(a, b1, acc[1]);
  acc[2] = wmma_bf16(a, b2, acc[2]);
  acc[3] = wmma_bf16(a, b3, acc[3]);

  // C/D layout: vgpr i -> m = m0 + i + 8*(lane>>4); n = n0 + (lane&15)
  const int mhi = (lane >> 4) * 8;
  const int nc  = lane & 15;
#pragma unroll
  for (int j = 0; j < 4; ++j) {
    int n = n0 + j * 16 + nc;
    float bv = bias ? bias[n] : 0.0f;
#pragma unroll
    for (int i = 0; i < 8; ++i) {
      int m = m0 + mhi + i;
      C[(long)m * N + n] = acc[j][i] + bv;
    }
  }
}

// ---------------------------------------------------------------------------
// One RNN timestep: h_t = tanh(ih_t + h_{t-1} @ w_hh^T)
// One wave per 16(n)-column tile; M = batch = 16 fits one WMMA exactly.
// h_seq layout: [(b*SEQ + t), HID] bf16.   Grid = HID/16, block = 32.
// ---------------------------------------------------------------------------
__global__ void k_rnn_step(const float* __restrict__ ih,
                           const __bf16* __restrict__ whhT,  // [HID, HID] K-major
                           __bf16* __restrict__ h_seq,
                           int t) {
  const int lane = threadIdx.x & 31;
  const int n0 = blockIdx.x * 16;

  v8f acc = {};
  if (t > 0) {
    const int b = lane & 15;                      // A row = batch index
    const int khalf = (lane >> 4) * 8;
    const __bf16* ap = h_seq + ((long)b * SEQ + (t - 1)) * HID + khalf;
    const __bf16* bp = whhT + (long)khalf * HID + n0 + (lane & 15);
    const long bstep = (long)32 * HID;

    v16bf a  = load_afrag(ap);
    v16bf bf = load_bfrag(bp, HID);
    for (int k0 = 32; k0 < HID; k0 += 32) {
      ap += 32;
      bp += bstep;
      __builtin_prefetch(bp + bstep, 0, 3);
      v16bf an  = load_afrag(ap);
      v16bf bfn = load_bfrag(bp, HID);
      acc = wmma_bf16(a, bf, acc);
      a = an; bf = bfn;
    }
    acc = wmma_bf16(a, bf, acc);
  }

  const int mhi = (lane >> 4) * 8;
  const int n = n0 + (lane & 15);
#pragma unroll
  for (int i = 0; i < 8; ++i) {
    int b = mhi + i;
    long idx = ((long)b * SEQ + t) * HID + n;
    float v = acc[i] + ih[idx];
    h_seq[idx] = (__bf16)tanhf(v);
  }
}

// ---------------------------------------------------------------------------
// Launch
// ---------------------------------------------------------------------------
extern "C" void kernel_launch(void* const* d_in, const int* in_sizes, int n_in,
                              void* d_out, int out_size, void* d_ws, size_t ws_size,
                              hipStream_t stream) {
  (void)in_sizes; (void)n_in; (void)out_size; (void)ws_size;

  const int*   x      = (const int*)d_in[0];
  const float* emb    = (const float*)d_in[1];
  const float* w_ih   = (const float*)d_in[2];
  const float* b_ih   = (const float*)d_in[3];
  const float* w_hh   = (const float*)d_in[4];
  const float* w_proj = (const float*)d_in[5];
  const float* b_proj = (const float*)d_in[6];
  float* out = (float*)d_out;

  // Workspace carve (each region 256B aligned)
  char* ws = (char*)d_ws;
  auto carve = [&](size_t bytes) -> char* {
    char* p = ws;
    ws += (bytes + 255) & ~(size_t)255;
    return p;
  };
  __bf16* e_bf   = (__bf16*)carve((size_t)MROWS * EMB * 2);   //  4 MB
  __bf16* wihT   = (__bf16*)carve((size_t)EMB * HID * 2);     //  1 MB
  __bf16* whhT   = (__bf16*)carve((size_t)HID * HID * 2);     //  2 MB
  __bf16* wprjT  = (__bf16*)carve((size_t)HID * VOCAB * 2);   // 62.5 MB
  float*  ih     = (float*) carve((size_t)MROWS * HID * 4);   // 16 MB
  __bf16* h_seq  = (__bf16*)carve((size_t)MROWS * HID * 2);   //  8 MB

  // 1) Embedding gather + bf16 convert
  {
    int total = MROWS * EMB;
    k_gather_embed<<<dim3((total + 255) / 256), 256, 0, stream>>>(x, emb, e_bf, total);
  }
  // 2) Transpose-convert weights to K-major bf16
  {
    long t1 = (long)HID * EMB;
    k_transpose_bf16<<<dim3((unsigned)((t1 + 255) / 256)), 256, 0, stream>>>(w_ih, wihT, HID, EMB);
    long t2 = (long)HID * HID;
    k_transpose_bf16<<<dim3((unsigned)((t2 + 255) / 256)), 256, 0, stream>>>(w_hh, whhT, HID, HID);
    long t3 = (long)VOCAB * HID;
    k_transpose_bf16<<<dim3((unsigned)((t3 + 255) / 256)), 256, 0, stream>>>(w_proj, wprjT, VOCAB, HID);
  }
  // 3) ih = e @ w_ih^T + b_ih     [4096 x 1024], K=512
  k_gemm_bf16_wmma<<<dim3(MROWS / 16, HID / 256), 128, 0, stream>>>(
      e_bf, wihT, b_ih, ih, MROWS, HID, EMB);

  // 4) Sequential recurrence: 256 per-timestep launches (graph-captured)
  for (int t = 0; t < SEQ; ++t) {
    k_rnn_step<<<dim3(HID / 16), 32, 0, stream>>>(ih, whhT, h_seq, t);
  }

  // 5) logits = h_seq @ w_proj^T + b_proj   [4096 x 32000], K=1024
  k_gemm_bf16_wmma<<<dim3(MROWS / 16, VOCAB / 256), 128, 0, stream>>>(
      h_seq, wprjT, b_proj, out, MROWS, VOCAB, HID);
}
